// MQACrossAttention_55310588837972
// MI455X (gfx1250) — matrix operand order repaired and checked by reading, not compiled, not run on hardware
//
#include <hip/hip_runtime.h>

// ---------------------------------------------------------------------------
// MQA cross-attention for MI455X (gfx1250, wave32, WMMA bf16 16x16x32)
// Shapes fixed: B=4 S=4096 T=512 D_MODEL=1024 D_TEXT=768 H=16 KVH=2 HD=64
// ---------------------------------------------------------------------------

typedef __attribute__((ext_vector_type(16))) __bf16 v16bf;
typedef __attribute__((ext_vector_type(8)))  float  v8f;

union BFrag { v16bf v; unsigned int u[8]; };

#define DEVI __device__ __forceinline__

DEVI v8f wmma_bf16(v16bf a, v16bf b, v8f c) {
  return __builtin_amdgcn_wmma_f32_16x16x32_bf16(
      /*neg_a=*/false, a, /*neg_b=*/false, b,
      /*c_mod=*/(short)0, c, /*reuse_a=*/false, /*reuse_b=*/false);
}

// Direct v_max_num_f32 (skip clang's canonicalization self-max pairs).
DEVI float vmax(float a, float b) {
  float d;
  asm("v_max_num_f32 %0, %1, %2" : "=v"(d) : "v"(a), "v"(b));
  return d;
}

// Max-reduce across the 16-lane DPP row (wave32 half) with fused
// v_max_num_f32_dpp row_ror:1/2/4/8 -- one VALU op per butterfly step.
DEVI float rowmax16(float x) {
  float t;
  asm("v_max_num_f32_dpp %0, %1, %1 row_ror:1 row_mask:0xf bank_mask:0xf"
      : "=v"(t) : "v"(x));
  asm("v_max_num_f32_dpp %0, %1, %1 row_ror:2 row_mask:0xf bank_mask:0xf"
      : "=v"(t) : "v"(t));
  asm("v_max_num_f32_dpp %0, %1, %1 row_ror:4 row_mask:0xf bank_mask:0xf"
      : "=v"(t) : "v"(t));
  asm("v_max_num_f32_dpp %0, %1, %1 row_ror:8 row_mask:0xf bank_mask:0xf"
      : "=v"(t) : "v"(t));
  return t;
}

// A fragment (16x32, M x K), wave32 layout:
// lanes 0-15: row=lane, K pairs {0..7,16..23}; lanes 16-31: row=lane-16, K pairs {8..15,24..31}
DEVI v16bf load_frag_a(const __bf16* base, int row0, int stride, int k0) {
  const int lane = threadIdx.x & 31;
  const int hh = lane >> 4;
  const int m  = lane & 15;
  const __bf16* rp = base + (size_t)(row0 + m) * stride + k0 + hh * 8;
  BFrag f;
#pragma unroll
  for (int i = 0; i < 4; ++i)
    f.u[i] = *(const unsigned int*)(rp + 2 * i);
#pragma unroll
  for (int i = 0; i < 4; ++i)
    f.u[4 + i] = *(const unsigned int*)(rp + 16 + 2 * i);
  return f.v;
}

// B fragment (32x16, K x N) read from a transposed [n][k] LDS tile:
// lanes 0-15: col=lane, K {0..7,16..23}; lanes 16-31: col=lane-16, K {8..15,24..31}
DEVI v16bf load_frag_b(const __bf16* baseT, int col0, int stride, int k0) {
  const int lane = threadIdx.x & 31;
  const int hh = lane >> 4;
  const int n  = lane & 15;
  const __bf16* cp = baseT + (size_t)(col0 + n) * stride + k0 + hh * 8;
  BFrag f;
#pragma unroll
  for (int j = 0; j < 8; ++j)
    f.u[j] = *(const unsigned int*)(cp + 2 * (j & 3) + (j >> 2) * 16);
  return f.v;
}

// ---------------------------------------------------------------------------
// f32 -> bf16 conversion, vectorized 8 elements/thread (sizes are mult. of 8)
// ---------------------------------------------------------------------------
__global__ __launch_bounds__(256) void cvt_kernel(const float* __restrict__ in,
                                                  __bf16* __restrict__ out, int n8) {
  for (int i = blockIdx.x * 256 + threadIdx.x; i < n8; i += gridDim.x * 256) {
    const float4 f0 = ((const float4*)in)[2 * i];
    const float4 f1 = ((const float4*)in)[2 * i + 1];
    __bf16 t[8] = {(__bf16)f0.x, (__bf16)f0.y, (__bf16)f0.z, (__bf16)f0.w,
                   (__bf16)f1.x, (__bf16)f1.y, (__bf16)f1.z, (__bf16)f1.w};
    ((uint4*)out)[i] = *(const uint4*)t;
  }
}

// ---------------------------------------------------------------------------
// LayerNorm over 768, bf16 output. One 256-thread block per row.
// ---------------------------------------------------------------------------
__global__ __launch_bounds__(256) void ln_kernel(const float* __restrict__ x,
                                                 const float* __restrict__ w,
                                                 const float* __restrict__ b,
                                                 __bf16* __restrict__ out) {
  __shared__ float red[256];
  const int row = blockIdx.x;
  const int tid = threadIdx.x;
  const float* xp = x + (size_t)row * 768;
  float v[3];
  float s = 0.f;
#pragma unroll
  for (int j = 0; j < 3; ++j) { v[j] = xp[tid + j * 256]; s += v[j]; }
  red[tid] = s; __syncthreads();
  for (int o = 128; o > 0; o >>= 1) {
    if (tid < o) red[tid] += red[tid + o];
    __syncthreads();
  }
  const float mu = red[0] * (1.0f / 768.0f);
  __syncthreads();
  float s2 = 0.f;
#pragma unroll
  for (int j = 0; j < 3; ++j) { const float d = v[j] - mu; s2 += d * d; }
  red[tid] = s2; __syncthreads();
  for (int o = 128; o > 0; o >>= 1) {
    if (tid < o) red[tid] += red[tid + o];
    __syncthreads();
  }
  const float r = rsqrtf(red[0] * (1.0f / 768.0f) + 1e-5f);
#pragma unroll
  for (int j = 0; j < 3; ++j) {
    const int c = tid + j * 256;
    out[(size_t)row * 768 + c] = (__bf16)((v[j] - mu) * r * w[c] + b[c]);
  }
}

// ---------------------------------------------------------------------------
// bf16 GEMM C[M,N] = A[M,K] x B[K,N], f32 accumulate.
// Block tile 128x64, 8 waves of 32x32, K-step 64, ping-pong double-buffered
// LDS (one barrier per K-tile, global loads overlap WMMA compute).
// EP=0: store bf16 C*cscale   EP=1: scatter K/V [B,KVH,T,HD]   EP=2: f32+bias
// ---------------------------------------------------------------------------
constexpr int GS_STR = 72;  // LDS stride for A and Bt tiles (elements)

template <int EP>
__global__ __launch_bounds__(256) void gemm_bf16(
    const __bf16* __restrict__ A, const __bf16* __restrict__ Bm,
    int M, int N, int K,
    __bf16* __restrict__ Cb, float cscale, float* __restrict__ Cf,
    const float* __restrict__ bias,
    __bf16* __restrict__ Kout, __bf16* __restrict__ Vout) {
  __shared__ __bf16 As[2][128 * GS_STR];
  __shared__ __bf16 Bs[2][64 * GS_STR];
  const int tid = threadIdx.x;
  const int wave = tid >> 5;
  const int lane = tid & 31;
  const int m0 = blockIdx.x * 128;
  const int n0 = blockIdx.y * 64;
  const int wm = (wave & 3) * 32;
  const int wn = (wave >> 2) * 32;

  uint4 ra[4], rb[2];
  auto gload = [&](int k0) {
#pragma unroll
    for (int it = 0; it < 4; ++it) {        // A tile 128x64: 1024 chunks of 8
      const int chunk = tid + it * 256;
      const int r = chunk >> 3, c = (chunk & 7) * 8;
      ra[it] = *(const uint4*)(A + (size_t)(m0 + r) * K + k0 + c);
    }
#pragma unroll
    for (int it = 0; it < 2; ++it) {        // B tile 64x64: 512 chunks of 8
      const int chunk = tid + it * 256;
      const int kk = chunk >> 3, c = (chunk & 7) * 8;
      rb[it] = *(const uint4*)(Bm + (size_t)(k0 + kk) * N + n0 + c);
    }
  };
  auto lstore = [&](int pp) {
    __bf16* Ap = As[pp];
    __bf16* Bp = Bs[pp];
#pragma unroll
    for (int it = 0; it < 4; ++it) {
      const int chunk = tid + it * 256;
      const int r = chunk >> 3, c = (chunk & 7) * 8;
      *(uint4*)&Ap[r * GS_STR + c] = ra[it];
    }
#pragma unroll
    for (int it = 0; it < 2; ++it) {        // transpose B -> [n][k]
      const int chunk = tid + it * 256;
      const int kk = chunk >> 3, c = (chunk & 7) * 8;
      const __bf16* pb = (const __bf16*)&rb[it];
#pragma unroll
      for (int j = 0; j < 8; ++j) Bp[(c + j) * GS_STR + kk] = pb[j];
    }
  };

  v8f acc[2][2] = {};
  gload(0);
  lstore(0);
  __syncthreads();
  int p = 0;
  for (int k0 = 0; k0 < K; k0 += 64) {
    const bool more = (k0 + 64) < K;
    if (more) gload(k0 + 64);               // next tile -> regs (overlaps WMMA)
    const __bf16* Ap = As[p];
    const __bf16* Bp = Bs[p];
    const v16bf a00 = load_frag_a(Ap, wm, GS_STR, 0);
    const v16bf a01 = load_frag_a(Ap, wm, GS_STR, 32);
    const v16bf a10 = load_frag_a(Ap, wm + 16, GS_STR, 0);
    const v16bf a11 = load_frag_a(Ap, wm + 16, GS_STR, 32);
    const v16bf b00 = load_frag_b(Bp, wn, GS_STR, 0);
    const v16bf b01 = load_frag_b(Bp, wn, GS_STR, 32);
    const v16bf b10 = load_frag_b(Bp, wn + 16, GS_STR, 0);
    const v16bf b11 = load_frag_b(Bp, wn + 16, GS_STR, 32);
    acc[0][0] = wmma_bf16(a00, b00, acc[0][0]);
    acc[0][0] = wmma_bf16(a01, b01, acc[0][0]);
    acc[0][1] = wmma_bf16(a00, b10, acc[0][1]);
    acc[0][1] = wmma_bf16(a01, b11, acc[0][1]);
    acc[1][0] = wmma_bf16(a10, b00, acc[1][0]);
    acc[1][0] = wmma_bf16(a11, b01, acc[1][0]);
    acc[1][1] = wmma_bf16(a10, b10, acc[1][1]);
    acc[1][1] = wmma_bf16(a11, b11, acc[1][1]);
    if (more) lstore(p ^ 1);                // fill other buffer
    __syncthreads();                        // single barrier per K-tile
    p ^= 1;
  }

  // Epilogue. C/D layout: lane = col (n%16), VGPR i = row i + 8*(lane>=16).
  const int hh = lane >> 4;
  const int nl = lane & 15;
#pragma unroll
  for (int mi = 0; mi < 2; ++mi)
#pragma unroll
    for (int ni = 0; ni < 2; ++ni)
#pragma unroll
      for (int i = 0; i < 8; ++i) {
        const int row = m0 + wm + mi * 16 + i + 8 * hh;
        const int col = n0 + wn + ni * 16 + nl;
        const float v = acc[mi][ni][i];
        if constexpr (EP == 0) {
          Cb[(size_t)row * N + col] = (__bf16)(v * cscale);
        } else if constexpr (EP == 1) {
          // row = b*T + t ; col < 128 -> K[g=col/64][d=col%64], else V
          const int bb = row >> 9;        // /512
          const int t  = row & 511;
          if (col < 128) {
            const int g = col >> 6, d = col & 63;
            Kout[((((size_t)bb * 2 + g) * 512) + t) * 64 + d] = (__bf16)v;
          } else {
            const int c2 = col - 128;
            const int g = c2 >> 6, d = c2 & 63;
            Vout[((((size_t)bb * 2 + g) * 512) + t) * 64 + d] = (__bf16)v;
          }
        } else {
          Cf[(size_t)row * N + col] = v + bias[col];
        }
      }
}

// ---------------------------------------------------------------------------
// Flash attention: block = (16 query rows, one (b,g)); 8 waves = 8 REP heads.
// Q[16x64] x K^T -> online softmax -> P x V, all via WMMA bf16.
// Row-sum accumulated on the matrix pipe via P x ones; row-max via fused
// DPP v_max_num_f32. K/V tiles ping-pong double-buffered: one barrier/T-tile.
// The 1/sqrt(HD) scale is pre-folded into Q by the Q-projection epilogue.
// ---------------------------------------------------------------------------
constexpr int KS_STR = 72;  // Ks [t][d] stride
constexpr int VS_STR = 40;  // Vs [d][t] stride
constexpr int PS_STR = 34;  // P staging stride

__global__ __launch_bounds__(256) void attn_kernel(
    const __bf16* __restrict__ Q, const __bf16* __restrict__ Kb,
    const __bf16* __restrict__ Vb, __bf16* __restrict__ Ob) {
  __shared__ __bf16 Ks[2][32 * KS_STR];
  __shared__ __bf16 Vs[2][64 * VS_STR];
  __shared__ __bf16 Ps[8 * 16 * PS_STR];

  const int tid = threadIdx.x;
  const int wave = tid >> 5;
  const int lane = tid & 31;
  const int hh = lane >> 4;
  const int nl = lane & 15;

  const int qbase = blockIdx.x * 16;        // 0..4095 step 16
  const int bg = blockIdx.y;                // 0..7
  const int b = bg >> 1, g = bg & 1;
  const int h = g * 8 + wave;               // head = g*REP + r

  // Q A-fragments straight from global (two K-steps over HD=64)
  const __bf16* qrow = Q + ((size_t)(b * 4096 + qbase)) * 1024 + h * 64;
  const v16bf qa0 = load_frag_a(qrow, 0, 1024, 0);
  const v16bf qa1 = load_frag_a(qrow, 0, 1024, 32);

  // all-ones B fragment (bf16 1.0 = 0x3f80): every D column = row sum
  BFrag onesf;
#pragma unroll
  for (int j = 0; j < 8; ++j) onesf.u[j] = 0x3f803f80u;
  const v16bf vones = onesf.v;

  float m_run[8];
#pragma unroll
  for (int i = 0; i < 8; ++i) m_run[i] = -1e30f;
  v8f o[4] = {};
  v8f lacc = {};
  __bf16* Pw = Ps + wave * 16 * PS_STR;
  const size_t kvbase = (((size_t)b * 2 + g) * 512) * 64;

  uint4 rk, rv;
  const int lt = tid >> 3;                  // 0..31  (t within tile)
  const int ld = (tid & 7) * 8;             // 0..56  (d group)
  auto kvload = [&](int t0) {
    rk = *(const uint4*)(Kb + kvbase + (size_t)(t0 + lt) * 64 + ld);
    rv = *(const uint4*)(Vb + kvbase + (size_t)(t0 + lt) * 64 + ld);
  };
  auto kvstore = [&](int pp) {
    *(uint4*)&Ks[pp][lt * KS_STR + ld] = rk;
    const __bf16* pv = (const __bf16*)&rv;
#pragma unroll
    for (int j = 0; j < 8; ++j) Vs[pp][(ld + j) * VS_STR + lt] = pv[j];  // transpose
  };

  kvload(0);
  kvstore(0);
  __syncthreads();
  int p = 0;

  for (int t0 = 0; t0 < 512; t0 += 32) {
    const bool more = (t0 + 32) < 512;
    if (more) kvload(t0 + 32);              // next K/V tile -> regs

    // scores: S[16x32] = Q[16x64] x K^T[64x32]  (Ks is already [n=t][k=d])
    v8f s0 = {}, s1 = {};
    s0 = wmma_bf16(qa0, load_frag_b(Ks[p], 0, KS_STR, 0), s0);
    s0 = wmma_bf16(qa1, load_frag_b(Ks[p], 0, KS_STR, 32), s0);
    s1 = wmma_bf16(qa0, load_frag_b(Ks[p], 16, KS_STR, 0), s1);
    s1 = wmma_bf16(qa1, load_frag_b(Ks[p], 16, KS_STR, 32), s1);

    // online softmax: row = vgpr i + 8*half; DPP row covers the 16-lane half
    float alpha[8];
#pragma unroll
    for (int i = 0; i < 8; ++i) {
      const float a = s0[i];
      const float c = s1[i];
      const float tm = rowmax16(vmax(a, c));
      const float mn = vmax(m_run[i], tm);
      const float p0 = __expf(a - mn);
      const float p1 = __expf(c - mn);
      alpha[i] = __expf(m_run[i] - mn);
      m_run[i] = mn;
      const int mrow = i + 8 * hh;
      Pw[mrow * PS_STR + nl] = (__bf16)p0;        // t tile 0
      Pw[mrow * PS_STR + 16 + nl] = (__bf16)p1;   // t tile 1
    }
#pragma unroll
    for (int i = 0; i < 8; ++i) {
      lacc[i] = lacc[i] * alpha[i];
#pragma unroll
      for (int jn = 0; jn < 4; ++jn) o[jn][i] = o[jn][i] * alpha[i];
    }

    asm volatile("s_wait_dscnt 0" ::: "memory");   // P staged C->A relayout
    const v16bf pa = load_frag_a(Pw, 0, PS_STR, 0);
    o[0] = wmma_bf16(pa, load_frag_b(Vs[p], 0, VS_STR, 0), o[0]);
    o[1] = wmma_bf16(pa, load_frag_b(Vs[p], 16, VS_STR, 0), o[1]);
    o[2] = wmma_bf16(pa, load_frag_b(Vs[p], 32, VS_STR, 0), o[2]);
    o[3] = wmma_bf16(pa, load_frag_b(Vs[p], 48, VS_STR, 0), o[3]);
    lacc = wmma_bf16(pa, vones, lacc);             // row sums, every column

    if (more) kvstore(p ^ 1);               // fill other buffer
    __syncthreads();                        // single barrier per T-tile
    p ^= 1;
  }

#pragma unroll
  for (int i = 0; i < 8; ++i) {
    const float inv = 1.0f / lacc[i];
    const int row = b * 4096 + qbase + i + 8 * hh;
#pragma unroll
    for (int jn = 0; jn < 4; ++jn)
      Ob[(size_t)row * 1024 + h * 64 + jn * 16 + nl] = (__bf16)(o[jn][i] * inv);
  }
}

// ---------------------------------------------------------------------------
// Orchestration
// ---------------------------------------------------------------------------
extern "C" void kernel_launch(void* const* d_in, const int* in_sizes, int n_in,
                              void* d_out, int out_size, void* d_ws, size_t ws_size,
                              hipStream_t stream) {
  constexpr int Bb = 4, S = 4096, T = 512, DM = 1024, DT = 768, KVH = 2, HD = 64;
  (void)in_sizes; (void)n_in; (void)out_size; (void)ws_size;

  const float* x    = (const float*)d_in[0];
  const float* text = (const float*)d_in[1];
  const float* lnw  = (const float*)d_in[2];
  const float* lnb  = (const float*)d_in[3];
  const float* Wq   = (const float*)d_in[4];
  const float* Wkv  = (const float*)d_in[5];
  const float* Wout = (const float*)d_in[6];
  const float* bout = (const float*)d_in[7];
  float* out = (float*)d_out;

  char* ws = (char*)d_ws;
  size_t off = 0;
  auto take = [&](size_t nbytes) -> void* {
    void* p = ws + off;
    off += (nbytes + 255) & ~(size_t)255;
    return p;
  };
  __bf16* xb   = (__bf16*)take((size_t)Bb * S * DM * 2);
  __bf16* wqb  = (__bf16*)take((size_t)DM * DM * 2);
  __bf16* wkvb = (__bf16*)take((size_t)DT * 2 * KVH * HD * 2);
  __bf16* wob  = (__bf16*)take((size_t)DM * DM * 2);
  __bf16* tnb  = (__bf16*)take((size_t)Bb * T * DT * 2);
  __bf16* qb   = (__bf16*)take((size_t)Bb * S * DM * 2);
  __bf16* kb   = (__bf16*)take((size_t)Bb * KVH * T * HD * 2);
  __bf16* vb   = (__bf16*)take((size_t)Bb * KVH * T * HD * 2);
  __bf16* ob   = (__bf16*)take((size_t)Bb * S * DM * 2);

  auto cvt = [&](const float* src, __bf16* dst, int n) {
    const int n8 = n / 8;
    int grid = (n8 + 255) / 256;
    if (grid > 4096) grid = 4096;
    cvt_kernel<<<grid, 256, 0, stream>>>(src, dst, n8);
  };
  cvt(x, xb, Bb * S * DM);
  cvt(Wq, wqb, DM * DM);
  cvt(Wkv, wkvb, DT * 2 * KVH * HD);
  cvt(Wout, wob, DM * DM);

  ln_kernel<<<Bb * T, 256, 0, stream>>>(text, lnw, lnb, tnb);

  // KV projection: [2048,768] x [768,256] with K/V scatter epilogue
  gemm_bf16<1><<<dim3((Bb * T) / 128, (2 * KVH * HD) / 64), 256, 0, stream>>>(
      tnb, wkvb, Bb * T, 2 * KVH * HD, DT, nullptr, 1.0f, nullptr, nullptr, kb, vb);

  // Q projection: [16384,1024] x [1024,1024] -> bf16 Q, pre-scaled by 1/sqrt(HD)
  gemm_bf16<0><<<dim3((Bb * S) / 128, DM / 64), 256, 0, stream>>>(
      xb, wqb, Bb * S, DM, DM, qb, 0.125f, nullptr, nullptr, nullptr, nullptr);

  // attention
  attn_kernel<<<dim3(S / 16, Bb * KVH), 256, 0, stream>>>(qb, kb, vb, ob);

  // output projection + bias -> f32 out
  gemm_bf16<2><<<dim3((Bb * S) / 128, DM / 64), 256, 0, stream>>>(
      ob, wob, Bb * S, DM, DM, nullptr, 1.0f, out, bout, nullptr, nullptr);
}
